// MultiHeadAttention_9792525435373
// MI455X (gfx1250) — matrix operand order repaired and checked by reading, not compiled, not run on hardware
//
#include <hip/hip_runtime.h>
#include <math.h>

// ---------- types ----------
typedef __bf16 bf16_t;
typedef __attribute__((ext_vector_type(16))) __bf16 v16bf;
typedef __attribute__((ext_vector_type(8)))  float  v8f;
typedef __attribute__((ext_vector_type(4)))  unsigned int u32x4;
typedef __attribute__((ext_vector_type(8)))  unsigned int u32x8;
typedef __attribute__((ext_vector_type(2)))  unsigned int u32x2;
typedef __attribute__((ext_vector_type(4)))  float  f32x4;
typedef __attribute__((ext_vector_type(4)))  int    i32x4;

typedef __attribute__((address_space(1))) i32x4 gbl_i32x4;
typedef __attribute__((address_space(3))) i32x4 lds_i32x4;

union Frag {
    v16bf  v;
    u32x4  u[2];
    bf16_t e[16];
};

// ---------- problem constants ----------
static constexpr int BB   = 2;
static constexpr int TT   = 2048;
static constexpr int DD   = 1024;   // D_IN == D_OUT
static constexpr int HH   = 16;
static constexpr int HDIM = 64;
static constexpr size_t MT = (size_t)BB * TT;   // 4096 rows

// ---------- CDNA5 async global->LDS copy (ASYNCcnt path) ----------
__device__ __forceinline__ void async_b128(const void* g, const void* l) {
    gbl_i32x4* gp = (gbl_i32x4*)(unsigned long long)g;
    lds_i32x4* lp = (lds_i32x4*)(unsigned int)(unsigned long long)l;
#if __has_builtin(__builtin_amdgcn_global_load_async_to_lds_b128)
    __builtin_amdgcn_global_load_async_to_lds_b128(gp, lp, 0, 0);
#else
    asm volatile("global_load_async_to_lds_b128 %0, %1, off"
                 :: "v"(lp), "v"(gp) : "memory");
#endif
}

#if __has_builtin(__builtin_amdgcn_s_wait_asynccnt)
#define WAIT_ASYNC(n) __builtin_amdgcn_s_wait_asynccnt(n)
#else
#define WAIT_ASYNC(n) asm volatile("s_wait_asynccnt %0" :: "n"(n) : "memory")
#endif

#if __has_builtin(__builtin_amdgcn_s_wait_tensorcnt)
#define WAIT_TENSOR(n) __builtin_amdgcn_s_wait_tensorcnt(n)
#else
#define WAIT_TENSOR(n) asm volatile("s_wait_tensorcnt %0" :: "n"(n) : "memory")
#endif

// ---------- CDNA5 Tensor Data Mover: 2D tile (rows x 32 bf16) -> LDS ----
// D# per CDNA5 ISA: group0 = {ctrl, lds_addr, global_addr,
// type=2}; group1 = {flags/pad, tensor dims, tile dims, dim0 stride}.
// LDS padding: +4 DWORDs after every 16 DWORDs => 40-element row pitch.
__device__ __forceinline__ void tdm_load_tile(const bf16_t* g, const bf16_t* l,
                                              int rows, int stride_elems) {
    unsigned long long ga = (unsigned long long)g;
    u32x4 g0;
    g0[0] = 1u;                                        // count=1, user D#
    g0[1] = (unsigned)(unsigned long long)l;           // lds byte address
    g0[2] = (unsigned)ga;                              // global addr [31:0]
    g0[3] = (unsigned)(ga >> 32) | (2u << 30);         // addr [56:32] | type=2
    u32x8 g1;
    g1[0] = (1u << 16)      // data_size = 1 (2 bytes)
          | (1u << 20)      // pad_enable
          | (3u << 22)      // pad_interval: 16 DWORDs (64B row)
          | (3u << 25);     // pad_amount: 4 DWORDs (8 bf16)
    g1[1] = 32u << 16;                                 // tensor_dim0 = 32
    g1[2] = (unsigned)rows << 16;                      // tensor_dim1 = rows
    g1[3] = 32u << 16;                                 // tile_dim0 = 32
    g1[4] = (unsigned)rows;                            // tile_dim1 = rows
    g1[5] = (unsigned)stride_elems;                    // tensor_dim0_stride
    g1[6] = 0u;
    g1[7] = 0u;
    asm volatile("tensor_load_to_lds %0, %1" :: "s"(g0), "s"(g1) : "memory");
}

// =====================================================================
// fp32 -> bf16 elementwise convert (4 elems / thread)
// =====================================================================
__global__ void k_f32_to_bf16(const float* __restrict__ x,
                              bf16_t* __restrict__ y, int n4) {
    int i = blockIdx.x * blockDim.x + threadIdx.x;
    if (i >= n4) return;
    f32x4 f = reinterpret_cast<const f32x4*>(x)[i];
    union { bf16_t b[4]; u32x2 u; } o;
    o.b[0] = (bf16_t)f.x; o.b[1] = (bf16_t)f.y;
    o.b[2] = (bf16_t)f.z; o.b[3] = (bf16_t)f.w;
    reinterpret_cast<u32x2*>(y)[i] = o.u;
}

// =====================================================================
// fp32 [K][N] -> bf16 [N][K]  (weight transpose+convert, 32x32 tiles)
// block (32,8), grid (N/32, K/32)
// =====================================================================
__global__ void k_transpose_bf16(const float* __restrict__ W,
                                 bf16_t* __restrict__ Wt, int K, int N) {
    __shared__ float t[32][33];
    const int n0 = blockIdx.x * 32, k0 = blockIdx.y * 32;
    const int tx = threadIdx.x, ty = threadIdx.y;
#pragma unroll
    for (int i = 0; i < 4; ++i)
        t[ty + i * 8][tx] = W[(size_t)(k0 + ty + i * 8) * N + n0 + tx];
    __syncthreads();
#pragma unroll
    for (int i = 0; i < 4; ++i)
        Wt[(size_t)(n0 + ty + i * 8) * K + k0 + tx] = (bf16_t)t[tx][ty + i * 8];
}

// =====================================================================
// GEMM: C[M][N] = A[M][K] * B[K][N], A bf16 row-major, Bt = B^T bf16
// row-major [N][K]. 128 threads = 4 waves; 64x64 tile, BK=32.
// Double-buffered LDS filled by the Tensor Data Mover (TENSORcnt path);
// wave w computes rows [w*16, w*16+16) x 64 cols via 4 WMMA accumulators.
// =====================================================================
template <bool OUT_BF16>
__global__ void k_gemm_bf16(const bf16_t* __restrict__ A,
                            const bf16_t* __restrict__ Bt,
                            void* __restrict__ Cv, int M, int N, int K) {
    constexpr int LDA = 40;                       // 32 + 8 pad (matches TDM pad)
    __shared__ __align__(16) bf16_t As[2][64 * LDA];
    __shared__ __align__(16) bf16_t Bs[2][64 * LDA];

    const int tid  = threadIdx.x;
    const int wave = tid >> 5;
    const int lane = tid & 31;
    const int l15  = lane & 15;
    const int hi   = lane >> 4;                   // K-half select
    const int m0   = blockIdx.y * 64;
    const int n0   = blockIdx.x * 64;

    v8f acc[4];
#pragma unroll
    for (int s = 0; s < 4; ++s)
#pragma unroll
        for (int j = 0; j < 8; ++j) acc[s][j] = 0.f;

    // TDM fill of one K-tile into buffer `buf` (wave 0 drives the DMA)
    auto issue_tile = [&](int k0, int buf) {
        tdm_load_tile(A  + (size_t)m0 * K + k0, &As[buf][0], 64, K);
        tdm_load_tile(Bt + (size_t)n0 * K + k0, &Bs[buf][0], 64, K);
    };

    if (wave == 0) issue_tile(0, 0);
    const int nt = K >> 5;
    for (int t = 0; t < nt; ++t) {
        const int cur = t & 1;
        if (t + 1 < nt) {
            if (wave == 0) {
                issue_tile((t + 1) << 5, cur ^ 1);
                WAIT_TENSOR(2);        // tile t done (in-order), t+1 in flight
            }
        } else if (wave == 0) {
            WAIT_TENSOR(0);
        }
        __syncthreads();

        // A fragment: row = l15, K halves {0..7,16..23} / {8..15,24..31}
        Frag af;
        const bf16_t* ap = &As[cur][(wave * 16 + l15) * LDA + hi * 8];
        af.u[0] = *reinterpret_cast<const u32x4*>(ap);
        af.u[1] = *reinterpret_cast<const u32x4*>(ap + 16);

#pragma unroll
        for (int s = 0; s < 4; ++s) {
            Frag bfr;   // B fragment: col = l15, K halves {0..7,16..23} / {8..15,24..31}
            const bf16_t* bp = &Bs[cur][(s * 16 + l15) * LDA + hi * 8];
            bfr.u[0] = *reinterpret_cast<const u32x4*>(bp);
            bfr.u[1] = *reinterpret_cast<const u32x4*>(bp + 16);
            acc[s] = __builtin_amdgcn_wmma_f32_16x16x32_bf16(
                false, af.v, false, bfr.v, (short)0, acc[s], false, false);
        }
        __syncthreads();
    }

    // C/D layout: VGPR j -> row (hi ? j+8 : j), col = l15 (+16*s)
    const int row = m0 + wave * 16 + hi * 8;
#pragma unroll
    for (int s = 0; s < 4; ++s) {
        const int col = n0 + s * 16 + l15;
#pragma unroll
        for (int j = 0; j < 8; ++j) {
            if (OUT_BF16)
                reinterpret_cast<bf16_t*>(Cv)[(size_t)(row + j) * N + col] = (bf16_t)acc[s][j];
            else
                reinterpret_cast<float*>(Cv)[(size_t)(row + j) * N + col] = acc[s][j];
        }
    }
}

// =====================================================================
// Causal flash attention, one (b,h) per blockIdx.y, 64 queries per block
// (wave w owns 16 queries). Key blocks of 32; K tile async-DMA'd to LDS,
// V tile transposed through registers; online softmax; P restaged via
// LDS into WMMA A layout. Next tile prefetched into L2.
// =====================================================================
__global__ void k_flash_attn(const bf16_t* __restrict__ Q,
                             const bf16_t* __restrict__ Km,
                             const bf16_t* __restrict__ V,
                             bf16_t* __restrict__ O) {
    __shared__ __align__(16) bf16_t Ks[32 * 64];       // [key][d]
    __shared__ __align__(16) bf16_t Vt[64 * 32];       // [d][key]
    __shared__ __align__(16) bf16_t Ps[4][16 * 32];    // per-wave P staging

    const int tid  = threadIdx.x;
    const int wave = tid >> 5;
    const int lane = tid & 31;
    const int l15  = lane & 15;
    const int hi   = lane >> 4;

    const int bh = blockIdx.y;
    const int b  = bh >> 4;
    const int h  = bh & 15;
    const int q0 = blockIdx.x * 64;
    const int qw = q0 + wave * 16;

    const size_t base = (size_t)b * TT * DD + (size_t)h * HDIM;
    const bf16_t* Qh = Q  + base;
    const bf16_t* Kh = Km + base;
    const bf16_t* Vh = V  + base;

    // Q fragments (head_dim 64 = two K=32 chunks), loaded once
    Frag qa[2];
#pragma unroll
    for (int c = 0; c < 2; ++c) {
        const bf16_t* qp = Qh + (size_t)(qw + l15) * DD + c * 32 + hi * 8;
        qa[c].u[0] = *reinterpret_cast<const u32x4*>(qp);
        qa[c].u[1] = *reinterpret_cast<const u32x4*>(qp + 16);
    }

    v8f oacc[4];
    float mrow[8], lrow[8];
#pragma unroll
    for (int s = 0; s < 4; ++s)
#pragma unroll
        for (int j = 0; j < 8; ++j) oacc[s][j] = 0.f;
#pragma unroll
    for (int j = 0; j < 8; ++j) { mrow[j] = -1e30f; lrow[j] = 0.f; }

    const int nkb = (q0 >> 5) + 2;     // key blocks of 32 covering [0, q0+64)
    const int key = tid >> 2;          // 0..31  (coop-load row)
    const int dch = (tid & 3) * 16;    // 0/16/32/48

    for (int it = 0; it < nkb; ++it) {
        const int kb = it * 32;
        __syncthreads();               // previous tile reads done
        {   // K tile: async DMA straight into LDS (row-major)
            const bf16_t* gk = Kh + (size_t)(kb + key) * DD + dch;
            async_b128(gk,     &Ks[key * 64 + dch]);
            async_b128(gk + 8, &Ks[key * 64 + dch + 8]);
            // V tile: registers -> transposed LDS scatter
            Frag vf;
            const u32x4* gv = reinterpret_cast<const u32x4*>(Vh + (size_t)(kb + key) * DD + dch);
            vf.u[0] = gv[0]; vf.u[1] = gv[1];
#pragma unroll
            for (int j = 0; j < 16; ++j)
                Vt[(dch + j) * 32 + key] = vf.e[j];
        }
        if (it + 1 < nkb) {            // L2 prefetch of next tile
            __builtin_prefetch(Kh + (size_t)(kb + 32 + key) * DD + dch, 0, 0);
            __builtin_prefetch(Vh + (size_t)(kb + 32 + key) * DD + dch, 0, 0);
        }
        WAIT_ASYNC(0);
        __syncthreads();

        // ---- scores: S = Q * K^T  (two 16-key subtiles) ----
        v8f s0, s1;
#pragma unroll
        for (int j = 0; j < 8; ++j) { s0[j] = 0.f; s1[j] = 0.f; }
#pragma unroll
        for (int c = 0; c < 2; ++c) {
            Frag b0, b1;
            const bf16_t* bp0 = &Ks[l15 * 64 + c * 32 + hi * 8];
            b0.u[0] = *reinterpret_cast<const u32x4*>(bp0);
            b0.u[1] = *reinterpret_cast<const u32x4*>(bp0 + 16);
            s0 = __builtin_amdgcn_wmma_f32_16x16x32_bf16(
                false, qa[c].v, false, b0.v, (short)0, s0, false, false);
            const bf16_t* bp1 = &Ks[(16 + l15) * 64 + c * 32 + hi * 8];
            b1.u[0] = *reinterpret_cast<const u32x4*>(bp1);
            b1.u[1] = *reinterpret_cast<const u32x4*>(bp1 + 16);
            s1 = __builtin_amdgcn_wmma_f32_16x16x32_bf16(
                false, qa[c].v, false, b1.v, (short)0, s1, false, false);
        }

        // ---- scale 1/sqrt(64) + causal mask ----
#pragma unroll
        for (int j = 0; j < 8; ++j) {
            const int q   = qw + hi * 8 + j;
            const int kg0 = kb + l15;
            const int kg1 = kb + 16 + l15;
            s0[j] = (kg0 > q) ? -1e30f : s0[j] * 0.125f;
            s1[j] = (kg1 > q) ? -1e30f : s1[j] * 0.125f;
        }

        // ---- online softmax (row stats via 16-lane shfl reductions) ----
        float alpha[8], p0[8], p1[8];
#pragma unroll
        for (int j = 0; j < 8; ++j) {
            float v = fmaxf(s0[j], s1[j]);
#pragma unroll
            for (int off = 1; off < 16; off <<= 1)
                v = fmaxf(v, __shfl_xor(v, off, 32));
            const float mn = fmaxf(mrow[j], v);
            alpha[j] = __expf(mrow[j] - mn);
            mrow[j]  = mn;
            p0[j] = __expf(s0[j] - mn);
            p1[j] = __expf(s1[j] - mn);
            float su = p0[j] + p1[j];
#pragma unroll
            for (int off = 1; off < 16; off <<= 1)
                su += __shfl_xor(su, off, 32);
            lrow[j] = lrow[j] * alpha[j] + su;
        }
#pragma unroll
        for (int s = 0; s < 4; ++s)
#pragma unroll
            for (int j = 0; j < 8; ++j) oacc[s][j] *= alpha[j];

        // ---- restage P (C-layout -> A-layout) through LDS ----
#pragma unroll
        for (int j = 0; j < 8; ++j) {
            Ps[wave][(hi * 8 + j) * 32 + l15]      = (bf16_t)p0[j];
            Ps[wave][(hi * 8 + j) * 32 + 16 + l15] = (bf16_t)p1[j];
        }
        __syncthreads();
        Frag pa;
        const bf16_t* pp = &Ps[wave][l15 * 32 + hi * 8];
        pa.u[0] = *reinterpret_cast<const u32x4*>(pp);
        pa.u[1] = *reinterpret_cast<const u32x4*>(pp + 16);

        // ---- O += P * V ----
#pragma unroll
        for (int s = 0; s < 4; ++s) {
            Frag vb;
            const bf16_t* vp = &Vt[(s * 16 + l15) * 32 + hi * 8];
            vb.u[0] = *reinterpret_cast<const u32x4*>(vp);
            vb.u[1] = *reinterpret_cast<const u32x4*>(vp + 16);
            oacc[s] = __builtin_amdgcn_wmma_f32_16x16x32_bf16(
                false, pa.v, false, vb.v, (short)0, oacc[s], false, false);
        }
    }

    // ---- normalize + write ctx (bf16 row-major [b*T+t][h*64+d]) ----
#pragma unroll
    for (int j = 0; j < 8; ++j) {
        const float inv = 1.f / lrow[j];
        const int q = qw + hi * 8 + j;
        bf16_t* op = O + base + (size_t)q * DD;
#pragma unroll
        for (int s = 0; s < 4; ++s)
            op[s * 16 + l15] = (bf16_t)(oacc[s][j] * inv);
    }
}

// =====================================================================
// host launcher
// =====================================================================
extern "C" void kernel_launch(void* const* d_in, const int* in_sizes, int n_in,
                              void* d_out, int out_size, void* d_ws, size_t ws_size,
                              hipStream_t stream) {
    (void)in_sizes; (void)n_in; (void)out_size; (void)ws_size;
    const float* x  = (const float*)d_in[0];
    const float* Wq = (const float*)d_in[1];
    const float* Wk = (const float*)d_in[2];
    const float* Wv = (const float*)d_in[3];
    const float* Wo = (const float*)d_in[4];

    char* ws = (char*)d_ws;
    bf16_t* xb  = (bf16_t*)ws;  ws += MT * DD * sizeof(bf16_t);
    bf16_t* WqT = (bf16_t*)ws;  ws += (size_t)DD * DD * sizeof(bf16_t);
    bf16_t* WkT = (bf16_t*)ws;  ws += (size_t)DD * DD * sizeof(bf16_t);
    bf16_t* WvT = (bf16_t*)ws;  ws += (size_t)DD * DD * sizeof(bf16_t);
    bf16_t* WoT = (bf16_t*)ws;  ws += (size_t)DD * DD * sizeof(bf16_t);
    bf16_t* Qb  = (bf16_t*)ws;  ws += MT * DD * sizeof(bf16_t);
    bf16_t* Kb  = (bf16_t*)ws;  ws += MT * DD * sizeof(bf16_t);
    bf16_t* Vb  = (bf16_t*)ws;  ws += MT * DD * sizeof(bf16_t);
    bf16_t* Cb  = (bf16_t*)ws;  ws += MT * DD * sizeof(bf16_t);

    // 1) convert x to bf16
    {
        int n4 = (int)(MT * DD / 4);
        k_f32_to_bf16<<<(n4 + 255) / 256, 256, 0, stream>>>(x, xb, n4);
    }
    // 2) transpose+convert weights
    {
        dim3 tb(32, 8), tg(DD / 32, DD / 32);
        k_transpose_bf16<<<tg, tb, 0, stream>>>(Wq, WqT, DD, DD);
        k_transpose_bf16<<<tg, tb, 0, stream>>>(Wk, WkT, DD, DD);
        k_transpose_bf16<<<tg, tb, 0, stream>>>(Wv, WvT, DD, DD);
        k_transpose_bf16<<<tg, tb, 0, stream>>>(Wo, WoT, DD, DD);
    }
    // 3) QKV projections (bf16 out)
    {
        dim3 gb(128), gg(DD / 64, (unsigned)(MT / 64));
        k_gemm_bf16<true><<<gg, gb, 0, stream>>>(xb, WqT, Qb, (int)MT, DD, DD);
        k_gemm_bf16<true><<<gg, gb, 0, stream>>>(xb, WkT, Kb, (int)MT, DD, DD);
        k_gemm_bf16<true><<<gg, gb, 0, stream>>>(xb, WvT, Vb, (int)MT, DD, DD);
    }
    // 4) causal flash attention
    {
        dim3 ab(128), ag(TT / 64, BB * HH);
        k_flash_attn<<<ag, ab, 0, stream>>>(Qb, Kb, Vb, Cb);
    }
    // 5) output projection (fp32 out -> d_out)
    {
        dim3 gb(128), gg(DD / 64, (unsigned)(MT / 64));
        k_gemm_bf16<false><<<gg, gb, 0, stream>>>(Cb, WoT, (float*)d_out, (int)MT, DD, DD);
    }
}
